// MultiHeadAttentionBlock_26113401160418
// MI455X (gfx1250) — compile-verified, matrix operand order checked by reading, not run on hardware
//
#include <hip/hip_runtime.h>

#define BDIM 2
#define SDIM 2048
#define DDIM 1024
#define HDIM 16
#define HDHD 64
#define MDIM (BDIM * SDIM)

typedef __bf16 bf16;
typedef __attribute__((ext_vector_type(16))) __bf16 v16bf;
typedef __attribute__((ext_vector_type(8)))  float  v8f;

// ---- CDNA5 Tensor Data Mover availability (device pass only) ----
#if defined(__has_builtin)
#if __has_builtin(__builtin_amdgcn_tensor_load_to_lds) && __has_builtin(__builtin_amdgcn_s_wait_tensorcnt)
#define CDNA5_TDM 1
#endif
#endif
#ifndef CDNA5_TDM
#define CDNA5_TDM 0
#endif

#if CDNA5_TDM
typedef __attribute__((ext_vector_type(4))) unsigned int v4u;
typedef __attribute__((ext_vector_type(8))) int         v8i_t;
typedef __attribute__((ext_vector_type(4))) int         v4i_t;

// Issue one TDM 2D tile load: rows x cols bf16 elements, global row stride
// `stride` elements, LDS destination padded by `pad` codes per ISA D# spec.
// pad_int_code: threshold = (1<<code) 8-byte units per row before padding.
// pad_amt_code: (code+1) DWORDs of LDS padding inserted.
static __device__ __forceinline__ void tdm_load_2d(const bf16* gptr, bf16* lptr,
                                                   unsigned rows, unsigned cols,
                                                   unsigned stride,
                                                   unsigned pad_int_code,
                                                   unsigned pad_amt_code) {
  const unsigned long long ga = (unsigned long long)(size_t)gptr;
  const unsigned lds = (unsigned)(size_t)lptr;
  v4u g0;
  g0[0] = 1u;                                             // count=1 (valid), user mode
  g0[1] = lds;                                            // lds_addr (bytes)
  g0[2] = (unsigned)(ga & 0xffffffffu);                   // global_addr[31:0]
  g0[3] = (unsigned)((ga >> 32) & 0x01ffffffu) | (2u << 30); // global_addr[56:32], type=2
  v8i_t g1;
  g1[0] = (int)((1u << 16)                                // data_size = 2 bytes
              | (1u << 20)                                // pad_enable
              | (pad_int_code << 22) | (pad_amt_code << 25));
  g1[1] = (int)((cols & 0xffffu) << 16);                  // tensor_dim0[15:0] @63:48
  g1[2] = (int)(((cols >> 16) & 0xffffu)                  // tensor_dim0[31:16]
              | ((rows & 0xffffu) << 16));                // tensor_dim1[15:0] @95:80
  g1[3] = (int)(((rows >> 16) & 0xffffu)                  // tensor_dim1[31:16]
              | ((cols & 0xffffu) << 16));                // tile_dim0 @127:112
  g1[4] = (int)(rows & 0xffffu);                          // tile_dim1 (tile_dim2 = 0)
  g1[5] = (int)stride;                                    // tensor_dim0_stride[31:0]
  g1[6] = 0;                                              // stride hi, dim1_stride lo
  g1[7] = 0;
  const v4i_t z4 = {0, 0, 0, 0};
#if __clang_major__ >= 23
  const v8i_t z8 = {0, 0, 0, 0, 0, 0, 0, 0};
  __builtin_amdgcn_tensor_load_to_lds(g0, g1, z4, z4, z8, 0);
#else
  __builtin_amdgcn_tensor_load_to_lds(g0, g1, z4, z4, 0);
#endif
}
#endif  // CDNA5_TDM

static __device__ __forceinline__ v8f wmma_bf16(v16bf a, v16bf b, v8f c) {
  // 8 args: (neg_a, A, neg_b, B, c_mod, C, reuse_a, reuse_b)
  return __builtin_amdgcn_wmma_f32_16x16x32_bf16(false, a, false, b, (short)0, c, false, false);
}

// Fragment loader matching the ISA 16-bit A-matrix 16x32 layout (B mirrors with
// lane->N): lane<16 -> K {0..7,16..23}; lane>=16 -> K {8..15,24..31}.
static __device__ __forceinline__ v16bf load_frag(const bf16* tile, int row, int ldt, int lane) {
  const int hi = (lane >> 4) & 1;
  const int l  = lane & 15;
  const bf16* p = tile + (size_t)(row + l) * ldt + hi * 8;
  union { uint4 u[2]; v16bf v; } un;
  un.u[0] = *(const uint4*)(p);
  un.u[1] = *(const uint4*)(p + 16);
  return un.v;
}

// ---------------------------------------------------------------- weights->bf16
__global__ void wconv_kernel(const float* __restrict__ wq, const float* __restrict__ wk,
                             const float* __restrict__ wv, const float* __restrict__ wo,
                             bf16* __restrict__ dst) {
  const int total = 4 * DDIM * DDIM;
  for (int i = blockIdx.x * blockDim.x + threadIdx.x; i < total; i += gridDim.x * blockDim.x) {
    const int w = i >> 20;                 // DDIM*DDIM == 1<<20
    const int r = i & ((1 << 20) - 1);
    const float* src = (w == 0) ? wq : (w == 1) ? wk : (w == 2) ? wv : wo;
    dst[i] = (bf16)src[r];
  }
}

// ---------------------------------------------------------------- layernorm
__global__ __launch_bounds__(256) void ln_kernel(const float* __restrict__ x,
                                                 const float* __restrict__ gamma,
                                                 const float* __restrict__ beta,
                                                 bf16* __restrict__ h) {
  const int row = blockIdx.x;
  const int t = threadIdx.x;
  const float4 v = ((const float4*)(x + (size_t)row * DDIM))[t];
  float s  = v.x + v.y + v.z + v.w;
  float ss = v.x * v.x + v.y * v.y + v.z * v.z + v.w * v.w;
  for (int off = 16; off > 0; off >>= 1) {
    s  += __shfl_down(s, off);
    ss += __shfl_down(ss, off);
  }
  __shared__ float ps[8], pss[8];
  __shared__ float smu, srstd;
  const int wid = t >> 5, lane = t & 31;
  if (lane == 0) { ps[wid] = s; pss[wid] = ss; }
  __syncthreads();
  if (t == 0) {
    float S = 0.f, SS = 0.f;
    for (int i = 0; i < 8; ++i) { S += ps[i]; SS += pss[i]; }
    const float mu  = S / (float)DDIM;
    const float var = SS / (float)DDIM - mu * mu;
    smu = mu; srstd = rsqrtf(var + 1e-5f);
  }
  __syncthreads();
  const float mu = smu, rstd = srstd;
  const float4 g  = ((const float4*)gamma)[t];
  const float4 be = ((const float4*)beta)[t];
  bf16* hp = h + (size_t)row * DDIM + t * 4;
  hp[0] = (bf16)((v.x - mu) * rstd * g.x + be.x);
  hp[1] = (bf16)((v.y - mu) * rstd * g.y + be.y);
  hp[2] = (bf16)((v.z - mu) * rstd * g.z + be.z);
  hp[3] = (bf16)((v.w - mu) * rstd * g.w + be.w);
}

// ---------------------------------------------------------------- 128x128 WMMA GEMM
// Computes Y[m,n] = sum_k A[m,k] * W[n,k] + bias[n].
// mode 0: z selects {q,k,v}; output bf16 in [B,H,S,HD] layout.
// mode 1: wo; output fp32 = x + Y into dout ([M,D]).
__global__ __launch_bounds__(256) void gemm128_kernel(
    const bf16* __restrict__ A, const bf16* __restrict__ wbase,
    const float* __restrict__ bq, const float* __restrict__ bk,
    const float* __restrict__ bv, const float* __restrict__ bo,
    const float* __restrict__ xres,
    bf16* __restrict__ qo, bf16* __restrict__ ko, bf16* __restrict__ vo,
    float* __restrict__ dout, int mode) {
#if CDNA5_TDM
  __shared__ bf16 As[2][128 * 40];
  __shared__ bf16 Bs[2][128 * 40];
#else
  __shared__ bf16 As[1][128 * 40];
  __shared__ bf16 Bs[1][128 * 40];
#endif
  const int t = threadIdx.x, lane = t & 31, wid = t >> 5;
  const int m0 = blockIdx.x * 128, n0 = blockIdx.y * 128;
  const int z = blockIdx.z;
  const bf16* W = wbase + (size_t)((mode == 1) ? 3 : z) * DDIM * DDIM;
  const float* bias = (mode == 1) ? bo : (z == 0 ? bq : (z == 1 ? bk : bv));

  const v8f vzero = {0.f, 0.f, 0.f, 0.f, 0.f, 0.f, 0.f, 0.f};
  v8f c[4][2];
#pragma unroll
  for (int i = 0; i < 4; ++i)
#pragma unroll
    for (int j = 0; j < 2; ++j) c[i][j] = vzero;

  const int wm = (wid >> 2) * 64;   // wave M offset (2 rows of waves)
  const int wn = (wid & 3) * 32;    // wave N offset (4 cols of waves)

#if CDNA5_TDM
  if (t < 32) {  // wave 0: prologue tiles into buffer 0 (pad: 64B rows -> +16B)
    tdm_load_2d(A + (size_t)m0 * DDIM, As[0], 128, 32, DDIM, 3, 3);
    tdm_load_2d(W + (size_t)n0 * DDIM, Bs[0], 128, 32, DDIM, 3, 3);
  }
  int cur = 0;
  for (int k0 = 0; k0 < DDIM; k0 += 32) {
    if (t < 32) {
      if (k0 + 32 < DDIM) {  // issue next buffer, wait for current pair
        tdm_load_2d(A + (size_t)m0 * DDIM + (k0 + 32), As[cur ^ 1], 128, 32, DDIM, 3, 3);
        tdm_load_2d(W + (size_t)n0 * DDIM + (k0 + 32), Bs[cur ^ 1], 128, 32, DDIM, 3, 3);
        __builtin_amdgcn_s_wait_tensorcnt((short)2);
      } else {
        __builtin_amdgcn_s_wait_tensorcnt((short)0);
      }
    }
    __syncthreads();
    const v16bf bf0 = load_frag(Bs[cur], wn,      40, lane);
    const v16bf bf1 = load_frag(Bs[cur], wn + 16, 40, lane);
#pragma unroll
    for (int mi = 0; mi < 4; ++mi) {
      const v16bf af = load_frag(As[cur], wm + mi * 16, 40, lane);
      c[mi][0] = wmma_bf16(af, bf0, c[mi][0]);
      c[mi][1] = wmma_bf16(af, bf1, c[mi][1]);
    }
    __syncthreads();
    cur ^= 1;
  }
#else
  const int lrow = t >> 1, lseg = t & 1;
  for (int k0 = 0; k0 < DDIM; k0 += 32) {
    const uint4* ga = (const uint4*)(A + (size_t)(m0 + lrow) * DDIM + k0 + lseg * 16);
    const uint4* gb = (const uint4*)(W + (size_t)(n0 + lrow) * DDIM + k0 + lseg * 16);
    const uint4 a0 = ga[0], a1 = ga[1];
    const uint4 b0 = gb[0], b1 = gb[1];
    if (k0 + 32 < DDIM) {
      __builtin_prefetch(A + (size_t)(m0 + lrow) * DDIM + k0 + 32 + lseg * 16, 0, 3);
      __builtin_prefetch(W + (size_t)(n0 + lrow) * DDIM + k0 + 32 + lseg * 16, 0, 3);
    }
    uint4* la = (uint4*)(As[0] + lrow * 40 + lseg * 16);
    uint4* lb = (uint4*)(Bs[0] + lrow * 40 + lseg * 16);
    la[0] = a0; la[1] = a1;
    lb[0] = b0; lb[1] = b1;
    __syncthreads();
    const v16bf bf0 = load_frag(Bs[0], wn,      40, lane);
    const v16bf bf1 = load_frag(Bs[0], wn + 16, 40, lane);
#pragma unroll
    for (int mi = 0; mi < 4; ++mi) {
      const v16bf af = load_frag(As[0], wm + mi * 16, 40, lane);
      c[mi][0] = wmma_bf16(af, bf0, c[mi][0]);
      c[mi][1] = wmma_bf16(af, bf1, c[mi][1]);
    }
    __syncthreads();
  }
#endif

  // Epilogue. C layout: lane<16 -> N=lane, M=e ; lane>=16 -> N=lane-16, M=e+8.
  const int hiL = (lane >> 4) & 1, l15 = lane & 15;
  bf16* outp = (z == 0) ? qo : ((z == 1) ? ko : vo);
#pragma unroll
  for (int mi = 0; mi < 4; ++mi) {
#pragma unroll
    for (int ni = 0; ni < 2; ++ni) {
#pragma unroll
      for (int e = 0; e < 8; ++e) {
        const int m = m0 + wm + mi * 16 + hiL * 8 + e;
        const int n = n0 + wn + ni * 16 + l15;
        const float val = c[mi][ni][e] + bias[n];
        if (mode == 0) {
          const int b = m >> 11, s = m & (SDIM - 1);
          const int hh = n >> 6, d = n & 63;
          outp[(((size_t)b * HDIM + hh) * SDIM + s) * HDHD + d] = (bf16)val;
        } else {
          dout[(size_t)m * DDIM + n] = xres[(size_t)m * DDIM + n] + val;
        }
      }
    }
  }
}

// ---------------------------------------------------------------- QK^T upper-tri row sums
// denom[q] = q + sum_{k>=q} exp(s[q,k]),  eqq[q] = exp(s[q,q]),  s = (Q.K)/32
__global__ __launch_bounds__(256) void scores_denom_kernel(
    const bf16* __restrict__ qws, const bf16* __restrict__ kws,
    float* __restrict__ denom, float* __restrict__ eqq) {
  __shared__ bf16 Qs[128 * 72];
#if CDNA5_TDM
  __shared__ bf16 Ks[2][128 * 72];
#else
  __shared__ bf16 Ks[1][128 * 72];
#endif
  __shared__ float rowsum[128];
  const int bh = blockIdx.y;
  const int q0 = blockIdx.x * 128;
  const bf16* Qp = qws + (size_t)bh * SDIM * HDHD;
  const bf16* Kp = kws + (size_t)bh * SDIM * HDHD;
  const int t = threadIdx.x, lane = t & 31, wid = t >> 5;
  const int lrow = t >> 1, lseg = t & 1;

  if (t < 128) rowsum[t] = 0.f;
  const int wm = (wid >> 2) * 64, wn = (wid & 3) * 32;
  const int hiL = (lane >> 4) & 1, l15 = lane & 15;
  float acc[4][8];
#pragma unroll
  for (int i = 0; i < 4; ++i)
#pragma unroll
    for (int e = 0; e < 8; ++e) acc[i][e] = 0.f;
  const v8f vzero = {0.f, 0.f, 0.f, 0.f, 0.f, 0.f, 0.f, 0.f};

#if CDNA5_TDM
  if (t < 32) {  // wave 0: Q tile + first K tile (pad: 128B rows -> +16B)
    tdm_load_2d(Qp + (size_t)q0 * HDHD, Qs,    128, 64, HDHD, 4, 3);
    tdm_load_2d(Kp + (size_t)q0 * HDHD, Ks[0], 128, 64, HDHD, 4, 3);
  }
  int cur = 0;
#else
  { // stage Q block (128 x 64)
    const uint4* g = (const uint4*)(Qp + (size_t)(q0 + lrow) * HDHD + lseg * 32);
    uint4* l = (uint4*)(Qs + lrow * 72 + lseg * 32);
    l[0] = g[0]; l[1] = g[1]; l[2] = g[2]; l[3] = g[3];
  }
#endif

  for (int kt = q0; kt < SDIM; kt += 128) {   // upper triangle only
#if CDNA5_TDM
    if (t < 32) {
      if (kt + 128 < SDIM) {
        tdm_load_2d(Kp + (size_t)(kt + 128) * HDHD, Ks[cur ^ 1], 128, 64, HDHD, 4, 3);
        __builtin_amdgcn_s_wait_tensorcnt((short)1);
      } else {
        __builtin_amdgcn_s_wait_tensorcnt((short)0);
      }
    }
    __syncthreads();
    const bf16* Kt = Ks[cur];
#else
    __syncthreads();
    {
      const uint4* g = (const uint4*)(Kp + (size_t)(kt + lrow) * HDHD + lseg * 32);
      uint4* l = (uint4*)(Ks[0] + lrow * 72 + lseg * 32);
      l[0] = g[0]; l[1] = g[1]; l[2] = g[2]; l[3] = g[3];
    }
    __syncthreads();
    const bf16* Kt = Ks[0];
#endif
    const v16bf b00 = load_frag(Kt,      wn,      72, lane);
    const v16bf b01 = load_frag(Kt + 32, wn,      72, lane);
    const v16bf b10 = load_frag(Kt,      wn + 16, 72, lane);
    const v16bf b11 = load_frag(Kt + 32, wn + 16, 72, lane);
#pragma unroll
    for (int mi = 0; mi < 4; ++mi) {
      const v16bf a0 = load_frag(Qs,      wm + mi * 16, 72, lane);
      const v16bf a1 = load_frag(Qs + 32, wm + mi * 16, 72, lane);
#pragma unroll
      for (int ni = 0; ni < 2; ++ni) {
        v8f cc = vzero;
        cc = wmma_bf16(a0, ni ? b10 : b00, cc);
        cc = wmma_bf16(a1, ni ? b11 : b01, cc);
        const int kg = kt + wn + ni * 16 + l15;
#pragma unroll
        for (int e = 0; e < 8; ++e) {
          const int qg = q0 + wm + mi * 16 + hiL * 8 + e;
          if (kg >= qg) {
            const float ev = __expf(cc[e] * 0.03125f);   // 1/sqrt(1024)
            acc[mi][e] += ev;
            if (kg == qg) eqq[(size_t)bh * SDIM + qg] = ev;
          }
        }
      }
    }
#if CDNA5_TDM
    __syncthreads();
    cur ^= 1;
#endif
  }

  // reduce across the 16 lanes sharing each C row, then across waves via LDS atomics
#pragma unroll
  for (int mi = 0; mi < 4; ++mi) {
#pragma unroll
    for (int e = 0; e < 8; ++e) {
      float v = acc[mi][e];
      v += __shfl_xor(v, 1);
      v += __shfl_xor(v, 2);
      v += __shfl_xor(v, 4);
      v += __shfl_xor(v, 8);
      if (l15 == 0) atomicAdd(&rowsum[wm + mi * 16 + hiL * 8 + e], v);
    }
  }
  __syncthreads();
  if (t < 128) {
    const int q = q0 + t;
    denom[(size_t)bh * SDIM + q] = rowsum[t] + (float)q;   // +q from exp(0) below-diag entries
  }
}

// ---------------------------------------------------------------- V prefix scan
// attn[b][s][h*64+d] = (sum_{k<s} V[k][d] + eqq[s]*V[s][d]) / denom[s]
__global__ __launch_bounds__(256) void scan_kernel(const bf16* __restrict__ vws,
                                                   const float* __restrict__ denom,
                                                   const float* __restrict__ eqq,
                                                   bf16* __restrict__ attn) {
  const int bh = blockIdx.x;
  const int b = bh >> 4, hh = bh & 15;
  const bf16* Vp = vws + (size_t)bh * SDIM * HDHD;
  const float* dn = denom + (size_t)bh * SDIM;
  const float* eq = eqq + (size_t)bh * SDIM;
  const int t = threadIdx.x;
  const int d = t & 63, g = t >> 6;        // 4 groups of 64 columns
  const int SEG = SDIM / 4;
  __shared__ float part[4][64];
  float s = 0.f;
  for (int q = g * SEG; q < (g + 1) * SEG; ++q) s += (float)Vp[(size_t)q * HDHD + d];
  part[g][d] = s;
  __syncthreads();
  float run = 0.f;
  for (int gg = 0; gg < g; ++gg) run += part[gg][d];
  bf16* ap = attn + (size_t)b * SDIM * DDIM + (size_t)hh * HDHD + d;
  for (int q = g * SEG; q < (g + 1) * SEG; ++q) {
    const float vq = (float)Vp[(size_t)q * HDHD + d];
    ap[(size_t)q * DDIM] = (bf16)((run + eq[q] * vq) / dn[q]);
    run += vq;
  }
}

// ---------------------------------------------------------------- launch
extern "C" void kernel_launch(void* const* d_in, const int* in_sizes, int n_in,
                              void* d_out, int out_size, void* d_ws, size_t ws_size,
                              hipStream_t stream) {
  const float* x     = (const float*)d_in[0];
  const float* wq    = (const float*)d_in[1];
  const float* bq    = (const float*)d_in[2];
  const float* wk    = (const float*)d_in[3];
  const float* bk    = (const float*)d_in[4];
  const float* wv    = (const float*)d_in[5];
  const float* bv    = (const float*)d_in[6];
  const float* wo    = (const float*)d_in[7];
  const float* bo    = (const float*)d_in[8];
  const float* gamma = (const float*)d_in[9];
  const float* beta  = (const float*)d_in[10];

  char* ws = (char*)d_ws;
  const size_t SZ_W   = (size_t)4 * DDIM * DDIM * sizeof(bf16);      // 8 MB
  const size_t SZ_MD  = (size_t)MDIM * DDIM * sizeof(bf16);          // 8 MB
  const size_t SZ_BHS = (size_t)BDIM * HDIM * SDIM * sizeof(float);  // 256 KB
  bf16*  wbf  = (bf16*)(ws);
  bf16*  hbf  = (bf16*)(ws + SZ_W);
  bf16*  qws  = (bf16*)(ws + SZ_W + 1 * SZ_MD);
  bf16*  kws  = (bf16*)(ws + SZ_W + 2 * SZ_MD);
  bf16*  vws  = (bf16*)(ws + SZ_W + 3 * SZ_MD);
  float* den  = (float*)(ws + SZ_W + 4 * SZ_MD);
  float* eqq  = (float*)(ws + SZ_W + 4 * SZ_MD + SZ_BHS);
  bf16*  attn = (bf16*)(ws + SZ_W + 4 * SZ_MD + 2 * SZ_BHS);
  float* out  = (float*)d_out;

  wconv_kernel<<<4096, 256, 0, stream>>>(wq, wk, wv, wo, wbf);
  ln_kernel<<<MDIM, 256, 0, stream>>>(x, gamma, beta, hbf);
  gemm128_kernel<<<dim3(MDIM / 128, DDIM / 128, 3), 256, 0, stream>>>(
      hbf, wbf, bq, bk, bv, bo, x, qws, kws, vws, out, 0);
  scores_denom_kernel<<<dim3(SDIM / 128, BDIM * HDIM), 256, 0, stream>>>(qws, kws, den, eqq);
  scan_kernel<<<BDIM * HDIM, 256, 0, stream>>>(vws, den, eqq, attn);
  gemm128_kernel<<<dim3(MDIM / 128, DDIM / 128, 1), 256, 0, stream>>>(
      attn, wbf, bq, bk, bv, bo, x, qws, kws, vws, out, 1);
}
